// TinyBlock_50843822850348
// MI455X (gfx1250) — matrix operand order repaired and checked by reading, not compiled, not run on hardware
//
#include <hip/hip_runtime.h>

typedef unsigned short u16;
typedef unsigned int   u32;
typedef unsigned long long u64;
typedef __attribute__((ext_vector_type(16))) __bf16 v16bf;
typedef __attribute__((ext_vector_type(8)))  float  v8f;
typedef __attribute__((ext_vector_type(4)))  u32    uivec4;

#define HID 1024
#define SEQ 2048
#define BATCH 2
#define NHEADS 16
#define HDIM 64
#define MROWS (BATCH * SEQ)   // 4096

union Frag16 {
    uivec4 u[2];
    v16bf  v;
    u16    s[16];
};

__device__ __forceinline__ u16 f2bf(float f) {
    u32 u = __float_as_uint(f);
    u += 0x7FFFu + ((u >> 16) & 1u);   // round-to-nearest-even
    return (u16)(u >> 16);
}

// xor-shuffle within 16-lane row via DPP ROW_XMASK (VALU, no LDS traffic).
#if __has_builtin(__builtin_amdgcn_update_dpp)
#define DPP_XOR(v, mask)                                                        \
    __uint_as_float(__builtin_amdgcn_update_dpp(                                \
        0u, __float_as_uint(v), 0x160 | (mask), 0xf, 0xf, true))
#else
#define DPP_XOR(v, mask) __shfl_xor((v), (mask), 32)
#endif

__device__ __forceinline__ float red_max16(float v) {
    v = fmaxf(v, DPP_XOR(v, 1));
    v = fmaxf(v, DPP_XOR(v, 2));
    v = fmaxf(v, DPP_XOR(v, 4));
    v = fmaxf(v, DPP_XOR(v, 8));
    return v;
}
__device__ __forceinline__ float red_sum16(float v) {
    v += DPP_XOR(v, 1);
    v += DPP_XOR(v, 2);
    v += DPP_XOR(v, 4);
    v += DPP_XOR(v, 8);
    return v;
}

// ---------------------------------------------------------------- TDM tile DMA
// 2D tile (tile_d1 rows x tile_d0 bf16 elems, row stride = stride_elems) from
// global to LDS via the Tensor Data Mover, D# built per ISA ch.8 (data_size=2B,
// count=1, type=2, no pad/iterate/multicast). Tracked on TENSORcnt.
// This toolchain exposes the 6-arg builtin:
//   (uint32x4 g0, int32x8 g1, int32x4 g2, int32x4 g3, int32x8 extra, i32 cpol)
#if __has_builtin(__builtin_amdgcn_tensor_load_to_lds) && \
    __has_builtin(__builtin_amdgcn_s_wait_tensorcnt)
#define USE_TDM 1
typedef __attribute__((ext_vector_type(4))) u32 v4u_t;
typedef __attribute__((ext_vector_type(8))) int v8i_t;
typedef __attribute__((ext_vector_type(4))) int v4i_t;

__device__ __forceinline__ void tdm_load_2d(const void* gptr, u32 lds_addr,
                                            u32 tile_d0, u32 tile_d1,
                                            u32 tensor_d0, u32 tensor_d1,
                                            u32 stride_elems) {
    u64 ga = (u64)(size_t)gptr;
    v4u_t g0;
    g0.x = 1u;                                            // count=1 (valid, user)
    g0.y = lds_addr;                                      // LDS byte address
    g0.z = (u32)ga;                                       // global_addr[31:0]
    g0.w = (u32)((ga >> 32) & 0x1FFFFFFu) | (2u << 30);   // global_addr[56:32] | type=2
    v8i_t g1;
    g1[0] = (int)(1u << 16);                              // data_size=1 -> 2 bytes
    g1[1] = (int)((tensor_d0 & 0xFFFFu) << 16);           // tensor_dim0[15:0]
    g1[2] = (int)(((tensor_d0 >> 16) & 0xFFFFu) |
                  ((tensor_d1 & 0xFFFFu) << 16));         // dim0[31:16] | dim1[15:0]
    g1[3] = (int)(((tensor_d1 >> 16) & 0xFFFFu) |
                  ((tile_d0 & 0xFFFFu) << 16));           // dim1[31:16] | tile_dim0
    g1[4] = (int)(tile_d1 & 0xFFFFu);                     // tile_dim1, tile_dim2=0 (2D)
    g1[5] = (int)stride_elems;                            // tensor_dim0_stride[31:0]
    g1[6] = 0;
    g1[7] = 0;
    v4i_t z4 = {0, 0, 0, 0};
    v8i_t z8 = {0, 0, 0, 0, 0, 0, 0, 0};
    __builtin_amdgcn_tensor_load_to_lds(g0, g1, z4, z4, z8, 0);
}
#else
#define USE_TDM 0
#endif

// ---------------------------------------------------------------- cast fp32 -> bf16
__global__ __launch_bounds__(256) void cast_bf16(const float* __restrict__ in,
                                                 u16* __restrict__ out, int n) {
    int i = blockIdx.x * 256 + threadIdx.x;
    if (i < n) out[i] = f2bf(in[i]);
}

// ---------------------------------------------------------------- tiled bf16 WMMA GEMM
// C[M,N] = A[M,K] @ B[K,N], A/B bf16 row-major, f32 accumulate.
// Tile: BM=64, BN=128, BK=32. 256 threads = 8 waves; wave w owns N strip w*16,
// and 4 M-subtiles of 16 -> 4 v_wmma_f32_16x16x32_bf16 per K step.
// A tile is DMA'd into LDS by the Tensor Data Mover (wave 0, TENSORcnt);
// B tile is staged transposed by all threads.
#define BM 64
#define BN 128
#define BK 32

template <bool RELU, bool OUT_BF16>
__global__ __launch_bounds__(256) void gemm_bf16(const u16* __restrict__ A,
                                                 const u16* __restrict__ Bm,
                                                 u16* __restrict__ Cb,
                                                 float* __restrict__ Cf,
                                                 int Ndim, int Kdim) {
    __shared__ u16 As[BM][BK];   // row-major A tile
    __shared__ u16 Bs[BN][BK];   // B tile stored transposed: Bs[n][k]

    const int tid  = threadIdx.x;
    const int wave = tid >> 5;
    const int lane = tid & 31;
    const int lm   = lane & 15;
    const int kh   = lane >> 4;
    const int m0   = blockIdx.x * BM;
    const int n0   = blockIdx.y * BN;

    v8f acc[4] = {};

    for (int k0 = 0; k0 < Kdim; k0 += BK) {
#if USE_TDM
        if (tid < 32)   // wave 0 issues the A-tile DMA (EXEC ignored by TDM)
            tdm_load_2d(A + (size_t)m0 * Kdim + k0, (u32)(size_t)&As[0][0],
                        BK, BM, (u32)Kdim, (u32)BM, (u32)Kdim);
#else
        {
            int r = tid >> 2;
            int c = (tid & 3) * 8;
            *(uivec4*)&As[r][c] =
                *(const uivec4*)(A + (size_t)(m0 + r) * Kdim + k0 + c);
        }
#endif
        // B tile: 32x128 bf16, 16 contiguous elems per thread, written transposed
        {
            int r = tid >> 3;          // K row 0..31
            int c = (tid & 7) * 16;    // N col 0..112
            const u16* src = Bm + (size_t)(k0 + r) * Ndim + n0 + c;
#pragma unroll
            for (int i = 0; i < 16; i++) Bs[c + i][r] = src[i];
        }
#if USE_TDM
        if (tid < 32) __builtin_amdgcn_s_wait_tensorcnt((short)0);
#endif
        __syncthreads();

        if (k0 + BK < Kdim)  // emits global_prefetch_b8 for next B tile
            __builtin_prefetch(Bm + (size_t)(k0 + BK + (tid >> 3)) * Ndim + n0, 0, 1);

        // B fragment: lane = column n, elems 0-7: K=kh*8+i, elems 8-15: K=16+kh*8+i
        Frag16 bf;
        bf.u[0] = *(const uivec4*)&Bs[wave * 16 + lm][kh * 8];
        bf.u[1] = *(const uivec4*)&Bs[wave * 16 + lm][16 + kh * 8];

#pragma unroll
        for (int mt = 0; mt < 4; mt++) {
            // A fragment: lane = row m, elems 0-7: K=kh*8+i, elems 8-15: K=16+kh*8+i
            Frag16 af;
            af.u[0] = *(const uivec4*)&As[mt * 16 + lm][kh * 8];
            af.u[1] = *(const uivec4*)&As[mt * 16 + lm][16 + kh * 8];
            acc[mt] = __builtin_amdgcn_wmma_f32_16x16x32_bf16(
                false, af.v, false, bf.v, (short)0, acc[mt], false, false);
        }
        __syncthreads();
    }

    // C layout: lane = N=lm, VGPR r -> M = r + 8*kh
#pragma unroll
    for (int mt = 0; mt < 4; mt++) {
#pragma unroll
        for (int r = 0; r < 8; r++) {
            size_t row = m0 + mt * 16 + r + 8 * kh;
            size_t col = n0 + wave * 16 + lm;
            float v = acc[mt][r];
            if (RELU) v = v > 0.0f ? v : 0.0f;
            if (OUT_BF16) Cb[row * Ndim + col] = f2bf(v);
            else          Cf[row * Ndim + col] = v;
        }
    }
}

// ---------------------------------------------------------------- flash attention
// qkv: [4096, 3072] bf16 (cols 0..1023 = Q, 1024.. = K, 2048.. = V; col = h*64+d)
// y:   [4096, 1024] bf16.
// One block = one (b,h), 4 waves = 4 consecutive 16-query tiles. Per 32-key
// step: K tile DMA'd row-major into LDS by the TDM, V tile staged transposed
// by all threads; both shared by the 4 waves as contiguous 16B LDS fragments.
__global__ __launch_bounds__(128) void attn_kernel(const u16* __restrict__ qkv,
                                                   u16* __restrict__ y) {
    __shared__ u16 Ks[32][64];        // K[key][d]      4KB
    __shared__ u16 Vt[64][32];        // V^T[d][key]    4KB
    __shared__ u16 Plds[4][16][32];   // per-wave P staging (C->A layout) 4KB

    const int tid  = threadIdx.x;
    const int wave = tid >> 5;
    const int lane = tid & 31;
    const int lm   = lane & 15;
    const int kh   = lane >> 4;

    const int qblk = blockIdx.x & 31;          // 32 blocks per (b,h)
    const int h    = (blockIdx.x >> 5) & (NHEADS - 1);
    const int b    = blockIdx.x >> 9;
    const int q0   = (qblk * 4 + wave) * 16;

    const size_t rs = 3 * HID;                 // 3072
    const u16* qb = qkv + (size_t)(b * SEQ) * rs + h * HDIM;
    const u16* kb = qb + HID;
    const u16* vb = qb + 2 * HID;

    // Q fragments (d split 0..31 / 32..63), loaded once per wave
    Frag16 qf[2];
#pragma unroll
    for (int s = 0; s < 2; s++) {
        const u16* p = qb + (size_t)(q0 + lm) * rs + s * 32;
        qf[s].u[0] = *(const uivec4*)(p + kh * 8);
        qf[s].u[1] = *(const uivec4*)(p + 16 + kh * 8);
    }

    float mi[8], li[8];
#pragma unroll
    for (int r = 0; r < 8; r++) { mi[r] = -1e30f; li[r] = 0.0f; }
    v8f oacc[4] = {};
    const float scale = 0.125f;   // 1/sqrt(64)

    for (int kbase = 0; kbase < SEQ; kbase += 32) {
        // ---- staging: TDM for K (wave 0), manual transpose for V (all threads)
#if USE_TDM
        if (tid < 32)
            tdm_load_2d(kb + (size_t)kbase * rs, (u32)(size_t)&Ks[0][0],
                        HDIM, 32, (u32)HDIM, (u32)SEQ, (u32)rs);
#endif
        {
            int r = tid >> 2;             // key row 0..31
            int c = (tid & 3) * 16;       // d col 0,16,32,48
#if !USE_TDM
            const u16* ksrc = kb + (size_t)(kbase + r) * rs + c;
            *(uivec4*)&Ks[r][c]     = *(const uivec4*)(ksrc);
            *(uivec4*)&Ks[r][c + 8] = *(const uivec4*)(ksrc + 8);
#endif
            const u16* vsrc = vb + (size_t)(kbase + r) * rs + c;
#pragma unroll
            for (int i = 0; i < 16; i++) Vt[c + i][r] = vsrc[i];
        }
#if USE_TDM
        if (tid < 32) __builtin_amdgcn_s_wait_tensorcnt((short)0);
#endif
        __syncthreads();

        if (kbase + 32 < SEQ)
            __builtin_prefetch(vb + (size_t)(kbase + 32 + (tid >> 2)) * rs, 0, 1);

        // ---- S = Q @ K^T for 32 keys -> two 16x16 C-frags
        v8f s01[2];
#pragma unroll
        for (int sub = 0; sub < 2; sub++) {
            v8f s = {};
#pragma unroll
            for (int ds = 0; ds < 2; ds++) {
                Frag16 kt;  // B-frag: lane n = key, elems 0-7: d = ds*32 + kh*8 + i, 8-15: d = ds*32 + 16 + kh*8 + i
                kt.u[0] = *(const uivec4*)&Ks[sub * 16 + lm][ds * 32 + kh * 8];
                kt.u[1] = *(const uivec4*)&Ks[sub * 16 + lm][ds * 32 + 16 + kh * 8];
                s = __builtin_amdgcn_wmma_f32_16x16x32_bf16(
                    false, qf[ds].v, false, kt.v, (short)0, s, false, false);
            }
            s01[sub] = s;
        }

        // ---- online softmax: per-lane stats cover rows r + 8*kh
        float p0[8], p1[8], corr[8];
#pragma unroll
        for (int r = 0; r < 8; r++) {
            float v  = red_max16(fmaxf(s01[0][r], s01[1][r]));
            float nm = fmaxf(mi[r], v * scale);
            corr[r] = __expf(mi[r] - nm);
            mi[r] = nm;
            p0[r] = __expf(s01[0][r] * scale - nm);
            p1[r] = __expf(s01[1][r] * scale - nm);
            li[r] = li[r] * corr[r] + red_sum16(p0[r] + p1[r]);
        }
#pragma unroll
        for (int dt = 0; dt < 4; dt++)
#pragma unroll
            for (int r = 0; r < 8; r++) oacc[dt][r] *= corr[r];

        // ---- P: C-layout -> A-layout via per-wave LDS (DS ops in-order in a wave)
#pragma unroll
        for (int r = 0; r < 8; r++) {
            Plds[wave][r + 8 * kh][lm]      = f2bf(p0[r]);
            Plds[wave][r + 8 * kh][16 + lm] = f2bf(p1[r]);
        }
        Frag16 pf;
        pf.u[0] = *(const uivec4*)&Plds[wave][lm][kh * 8];
        pf.u[1] = *(const uivec4*)&Plds[wave][lm][16 + kh * 8];

        // ---- O += P @ V (4 d-tiles of 16); B-frag from transposed V in LDS
#pragma unroll
        for (int dt = 0; dt < 4; dt++) {
            Frag16 vf;  // lane n = d col, elems 0-7: key kh*8 + i, 8-15: key 16 + kh*8 + i
            vf.u[0] = *(const uivec4*)&Vt[dt * 16 + lm][kh * 8];
            vf.u[1] = *(const uivec4*)&Vt[dt * 16 + lm][16 + kh * 8];
            oacc[dt] = __builtin_amdgcn_wmma_f32_16x16x32_bf16(
                false, pf.v, false, vf.v, (short)0, oacc[dt], false, false);
        }
        __syncthreads();
    }

    // epilogue: divide by row sums, store bf16 into [B*T, H] (col = h*64+d)
#pragma unroll
    for (int dt = 0; dt < 4; dt++) {
#pragma unroll
        for (int r = 0; r < 8; r++) {
            size_t row = (size_t)b * SEQ + q0 + r + 8 * kh;
            y[row * HID + h * HDIM + dt * 16 + lm] = f2bf(oacc[dt][r] / li[r]);
        }
    }
}

// ---------------------------------------------------------------- host launch
static constexpr size_t OFF_XB    = 0;                                   // 4096x1024 bf16 (also proj out)
static constexpr size_t OFF_WQKV  = OFF_XB    + 4096ull * 1024 * 2;      // 1024x3072 bf16
static constexpr size_t OFF_WPROJ = OFF_WQKV  + 1024ull * 3072 * 2;      // 1024x1024 bf16
static constexpr size_t OFF_WFF1  = OFF_WPROJ + 1024ull * 1024 * 2;      // 1024x4096 bf16
static constexpr size_t OFF_WFF2  = OFF_WFF1  + 1024ull * 4096 * 2;      // 4096x1024 bf16
static constexpr size_t OFF_QKV   = OFF_WFF2  + 4096ull * 1024 * 2;      // 4096x3072 bf16
static constexpr size_t OFF_Y     = OFF_QKV   + 4096ull * 3072 * 2;      // 4096x1024 bf16
static constexpr size_t OFF_FF1   = OFF_Y     + 4096ull * 1024 * 2;      // 4096x4096 bf16

extern "C" void kernel_launch(void* const* d_in, const int* in_sizes, int n_in,
                              void* d_out, int out_size, void* d_ws, size_t ws_size,
                              hipStream_t stream) {
    const float* x     = (const float*)d_in[0];
    const float* wqkv  = (const float*)d_in[1];
    const float* wproj = (const float*)d_in[2];
    const float* wff1  = (const float*)d_in[3];
    const float* wff2  = (const float*)d_in[4];
    float* out = (float*)d_out;
    char*  ws  = (char*)d_ws;

    u16* xb    = (u16*)(ws + OFF_XB);
    u16* wqkvb = (u16*)(ws + OFF_WQKV);
    u16* wprojb= (u16*)(ws + OFF_WPROJ);
    u16* wff1b = (u16*)(ws + OFF_WFF1);
    u16* wff2b = (u16*)(ws + OFF_WFF2);
    u16* qkvb  = (u16*)(ws + OFF_QKV);
    u16* yb    = (u16*)(ws + OFF_Y);
    u16* ff1b  = (u16*)(ws + OFF_FF1);
    u16* projb = xb;   // x bf16 is dead after the qkv GEMM; reuse for proj output

    auto cast = [&](const float* src, u16* dst, int n) {
        cast_bf16<<<(n + 255) / 256, 256, 0, stream>>>(src, dst, n);
    };
    cast(x,     xb,     MROWS * HID);
    cast(wqkv,  wqkvb,  HID * 3 * HID);
    cast(wproj, wprojb, HID * HID);
    cast(wff1,  wff1b,  HID * 4 * HID);
    cast(wff2,  wff2b,  4 * HID * HID);

    // qkv = x @ w_qkv                [4096,3072]
    gemm_bf16<false, true><<<dim3(MROWS / BM, 3 * HID / BN), 256, 0, stream>>>(
        xb, wqkvb, qkvb, nullptr, 3 * HID, HID);

    // attention                      [4096,1024]
    attn_kernel<<<BATCH * NHEADS * (SEQ / 16) / 4, 128, 0, stream>>>(qkvb, yb);

    // proj = y @ w_proj              [4096,1024]
    gemm_bf16<false, true><<<dim3(MROWS / BM, HID / BN), 256, 0, stream>>>(
        yb, wprojb, projb, nullptr, HID, HID);

    // ff1 = relu(proj @ w_ff1)       [4096,4096]
    gemm_bf16<true, true><<<dim3(MROWS / BM, 4 * HID / BN), 256, 0, stream>>>(
        projb, wff1b, ff1b, nullptr, 4 * HID, HID);

    // out = ff1 @ w_ff2  (f32)       [4096,1024]
    gemm_bf16<false, false><<<dim3(MROWS / BM, HID / BN), 256, 0, stream>>>(
        ff1b, wff2b, nullptr, out, HID, 4 * HID);
}
